// MambaOperator_20237885899506
// MI455X (gfx1250) — matrix operand order repaired and checked by reading, not compiled, not run on hardware
//
#include <hip/hip_runtime.h>
#include <hip/hip_bf16.h>

// ---------------- problem constants ----------------
#define BATCH   8
#define SEQ     2048
#define DM      768
#define DI      1536          // d_inner
#define NXZ     3072          // 2*d_inner
#define NPROJ   176           // DT_RANK + 2*D_STATE
#define DTRANK  48
#define DSTATE  64
#define BT      (BATCH*SEQ)   // 16384 rows

typedef __attribute__((ext_vector_type(16))) __bf16 v16bf;
typedef __attribute__((ext_vector_type(8)))  float  v8f;
typedef __attribute__((ext_vector_type(4)))  unsigned int v4u;
typedef __attribute__((ext_vector_type(8)))  int v8i;
typedef __attribute__((ext_vector_type(4)))  int v4i;

union FragBF { uint4 q[2]; v16bf v; };

#if defined(__gfx1250__) && __has_builtin(__builtin_amdgcn_tensor_load_to_lds)
#define USE_TDM 1
#else
#define USE_TDM 0
#endif

#if USE_TDM
// Issue a TDM 2D tile load: tile_m rows x tile_k elems of bf16, global row
// stride = row_stride elems, into contiguous LDS at lds_off.  Descriptor
// layout per CDNA5 ISA ch.8 (D# group0 + group1; groups 2/3 zero => 2D).
__device__ __forceinline__ void tdm_load_2d_bf16(const __bf16* gsrc, unsigned lds_off,
                                                 int tile_k, int tile_m,
                                                 long long row_stride) {
    unsigned long long ga = (unsigned long long)(uintptr_t)gsrc;
    v4u g0;
    g0.x = 1u;                                        // count=1 (valid), user mode
    g0.y = lds_off;                                   // lds_addr (bytes)
    g0.z = (unsigned)(ga & 0xffffffffu);              // global_addr[31:0]
    g0.w = (unsigned)((ga >> 32) & 0x01ffffffu)       // global_addr[56:32]
         | (2u << 30);                                // type = 2 ("image")
    unsigned d0 = (unsigned)tile_k;                   // tensor_dim0 (elems)
    unsigned d1 = (unsigned)tile_m;                   // tensor_dim1 (rows)
    unsigned long long s0 = (unsigned long long)row_stride;
    v8i g1;
    g1[0] = (int)(1u << 16);                          // wg_mask=0, data_size=1 (2B)
    g1[1] = (int)((d0 & 0xffffu) << 16);              // tensor_dim0[15:0]
    g1[2] = (int)(((d0 >> 16) & 0xffffu) | ((d1 & 0xffffu) << 16));
    g1[3] = (int)(((d1 >> 16) & 0xffffu) | ((unsigned)tile_k << 16));  // tile_dim0
    g1[4] = (int)((unsigned)tile_m & 0xffffu);        // tile_dim1; tile_dim2=0
    g1[5] = (int)(s0 & 0xffffffffu);                  // tensor_dim0_stride[31:0]
    g1[6] = (int)((s0 >> 32) & 0xffffu);              // stride[47:32]; dim1_stride=0
    g1[7] = 0;
    v4i z4 = {0, 0, 0, 0};
#if __clang_major__ >= 23
    v8i z8 = {0, 0, 0, 0, 0, 0, 0, 0};
    __builtin_amdgcn_tensor_load_to_lds(g0, g1, z4, z4, z8, 0);
#else
    __builtin_amdgcn_tensor_load_to_lds(g0, g1, z4, z4, 0);
#endif
}
#endif

// ---------------- conversion kernels ----------------
__global__ void x_mask_to_bf16(const float* __restrict__ x,
                               const unsigned char* __restrict__ mask,
                               __bf16* __restrict__ out, long long n) {
    long long i = (long long)blockIdx.x * 256 + threadIdx.x;
    if (i >= n) return;
    float m = mask[i / DM] ? 1.0f : 0.0f;
    out[i] = (__bf16)(x[i] * m);
}

__global__ void f32_to_bf16(const float* __restrict__ in,
                            __bf16* __restrict__ out, long long n) {
    long long i = (long long)blockIdx.x * 256 + threadIdx.x;
    if (i >= n) return;
    out[i] = (__bf16)in[i];
}

// ---------------- bf16 WMMA GEMM: C[M,N] = A[M,K] @ B[K,N] ----------------
// block: 256 thr = 8 waves; block tile 128x64; wave tile 32x32 (2x2 WMMA 16x16x32)
// A tile staged by the Tensor Data Mover (async DMA, TENSORcnt), B tile staged
// transposed by vector loads + packed b32 LDS stores.
__global__ __launch_bounds__(256)
void gemm_bf16_wmma(const __bf16* __restrict__ A, const __bf16* __restrict__ Bw,
                    float* __restrict__ C, int M, int N, int K) {
    __shared__ __attribute__((aligned(16))) __bf16 sA[128 * 32]; // row-major, stride 32
    __shared__ __attribute__((aligned(16))) __bf16 sB[64 * 32];  // transposed: [n][k]

    const int tid  = threadIdx.x;
    const int lane = tid & 31;
    const int wave = tid >> 5;
    const int wm   = wave >> 1;        // 0..3  (wave row)
    const int wn   = wave & 1;         // 0..1  (wave col)
    const int bm   = blockIdx.y * 128;
    const int bn   = blockIdx.x * 64;

    v8f acc[2][2] = {};

    const int mfrag = lane & 15;
    const int kAoff = (lane >> 4) * 8;    // A: K-subblock {0,8} then {16,24}
    const int kBoff = (lane >> 4) * 16;   // unused; B reads also use kAoff

    for (int k0 = 0; k0 < K; k0 += 32) {
        // ---- stage A tile (128 rows x 32 k) ----
#if USE_TDM
        if (wave == 0) {   // uniform branch: one TDM op per workgroup
            tdm_load_2d_bf16(A + (long long)bm * K + k0,
                             (unsigned)(uintptr_t)sA, 32, 128, (long long)K);
        }
#else
        {
            int arow = tid >> 1;              // 0..127
            int aseg = (tid & 1) * 16;        // 0 or 16
            const uint4* g = (const uint4*)(A + (long long)(bm + arow) * K + k0 + aseg);
            uint4 v0 = g[0];
            uint4 v1 = g[1];
            *(uint4*)&sA[arow * 32 + aseg]     = v0;
            *(uint4*)&sA[arow * 32 + aseg + 8] = v1;
        }
#endif
        // ---- stage B tile transposed (64n x 32k), vectorized ----
        // 128 units: 8 n-chunks (8 cols each) x 16 k-pairs.  Each active thread
        // loads 8 bf16 from rows k and k+1 (b128, coalesced: n-chunk fastest)
        // and writes 8 packed k-pairs as b32.  N is a multiple of 8, bn of 64,
        // so each 8-chunk is wholly in or out of bounds.
        if (tid < 128) {
            int nchunk = tid & 7;
            int kpair  = tid >> 3;            // 0..15
            int kk     = kpair * 2;
            int nn8    = nchunk * 8;
            union Pk { uint4 q; unsigned short us[8]; } r0 = {}, r1 = {};
            if (bn + nn8 < N) {
                const __bf16* gp = Bw + (long long)(k0 + kk) * N + bn + nn8;
                r0.q = *(const uint4*)gp;
                r1.q = *(const uint4*)(gp + N);
            }
            unsigned* sB32 = (unsigned*)sB;   // 16 dwords per n-row
            #pragma unroll
            for (int j = 0; j < 8; ++j)
                sB32[(nn8 + j) * 16 + kpair] =
                    (unsigned)r0.us[j] | ((unsigned)r1.us[j] << 16);
        }
        // prefetch next B tile rows into cache (global_prefetch_b8)
        if (k0 + 32 < K && tid < 32)
            __builtin_prefetch(Bw + (long long)(k0 + 32 + tid) * N + bn, 0, 1);
#if USE_TDM
        if (wave == 0)
            __builtin_amdgcn_s_wait_tensorcnt((short)0);   // A tile landed in LDS
#endif
        __syncthreads();

        FragBF a[2], b[2];
        #pragma unroll
        for (int i = 0; i < 2; ++i) {
            int m = wm * 32 + i * 16 + mfrag;
            a[i].q[0] = *(const uint4*)&sA[m * 32 + kAoff];        // K kAoff..+7
            a[i].q[1] = *(const uint4*)&sA[m * 32 + 16 + kAoff];   // K 16+kAoff..+7
        }
        #pragma unroll
        for (int j = 0; j < 2; ++j) {
            int n = wn * 32 + j * 16 + mfrag;
            b[j].q[0] = *(const uint4*)&sB[n * 32 + kAoff];        // K kAoff..+7
            b[j].q[1] = *(const uint4*)&sB[n * 32 + 16 + kAoff];   // K 16+kAoff..+7
        }
        #pragma unroll
        for (int i = 0; i < 2; ++i)
            #pragma unroll
            for (int j = 0; j < 2; ++j)
                acc[i][j] = __builtin_amdgcn_wmma_f32_16x16x32_bf16(
                    false, a[i].v, false, b[j].v, (short)0, acc[i][j], false, false);
        __syncthreads();
    }

    // ---- store D: element r -> (M = r + 8*(lane>=16), N = lane&15) ----
    const int nbase = bn + wn * 32 + (lane & 15);
    const int mbase = bm + wm * 32 + ((lane >> 4) ? 8 : 0);
    #pragma unroll
    for (int i = 0; i < 2; ++i)
        #pragma unroll
        for (int j = 0; j < 2; ++j) {
            int n = nbase + j * 16;
            if (n < N) {
                #pragma unroll
                for (int r = 0; r < 8; ++r)
                    C[(long long)(mbase + i * 16 + r) * N + n] = acc[i][j][r];
            }
        }
}

// ---------------- depthwise causal conv (K=4) + bias + SiLU ----------------
__global__ void conv_silu_kernel(const float* __restrict__ xz,
                                 const float* __restrict__ conv_w,
                                 const float* __restrict__ conv_b,
                                 float* __restrict__ u, __bf16* __restrict__ u_bf) {
    long long i = (long long)blockIdx.x * 256 + threadIdx.x;   // over BT*DI
    if (i >= (long long)BT * DI) return;
    int d       = (int)(i % DI);
    long long r = i / DI;          // global row b*SEQ + t
    int t       = (int)(r % SEQ);
    float acc = conv_b[d];
    #pragma unroll
    for (int j = 0; j < 4; ++j) {
        int tt = t - 3 + j;
        if (tt >= 0)
            acc += xz[(r - 3 + j) * NXZ + d] * conv_w[d * 4 + j];
    }
    float s = acc / (1.0f + __expf(-acc));   // SiLU
    u[i]    = s;
    u_bf[i] = (__bf16)s;
}

// ---------------- dt = softplus(proj[:, :48] @ W_dt + b_dt) ----------------
__global__ void dtproj_kernel(const float* __restrict__ proj,
                              const float* __restrict__ W_dt,
                              const float* __restrict__ b_dt,
                              float* __restrict__ dtout) {
    int d   = blockIdx.x * 256 + threadIdx.x;   // 0..1535
    int row = blockIdx.y;                       // 0..16383
    if (d >= DI) return;
    float acc = b_dt[d];
    const float* pr = proj + (long long)row * NPROJ;
    #pragma unroll 8
    for (int r = 0; r < DTRANK; ++r)
        acc += pr[r] * W_dt[(long long)r * DI + d];
    float sp = (acc > 20.0f) ? acc : __logf(1.0f + __expf(acc));
    dtout[(long long)row * DI + d] = sp;
}

// ---------------- selective scan ----------------
// block = 8 waves, one b; wave w owns channel d = blockIdx.x*8 + w.
// lane owns states {lane, lane+32}.  B/C staged in LDS per 64-step chunk,
// shared by all 8 waves (8x reuse).  y via cross-lane shfl_xor reduction.
#define SCAN_TS 64
__global__ __launch_bounds__(256)
void scan_kernel(const float* __restrict__ proj, const float* __restrict__ dt,
                 const float* __restrict__ u, const float* __restrict__ A_log,
                 float* __restrict__ y) {
    __shared__ float sBC[SCAN_TS][128];   // [t][0..63]=B, [64..127]=C  (32 KB)
    __shared__ float sDT[8][SCAN_TS];
    __shared__ float sU [8][SCAN_TS];

    const int lane = threadIdx.x & 31;
    const int wave = threadIdx.x >> 5;
    const int b    = blockIdx.y;
    const int d    = blockIdx.x * 8 + wave;

    const float A0 = -__expf(A_log[d * DSTATE + lane]);
    const float A1 = -__expf(A_log[d * DSTATE + 32 + lane]);
    float h0 = 0.0f, h1 = 0.0f;

    for (int t0 = 0; t0 < SEQ; t0 += SCAN_TS) {
        for (int idx = threadIdx.x; idx < SCAN_TS * 128; idx += 256) {
            int tl = idx >> 7, c = idx & 127;
            sBC[tl][c] = proj[(long long)(b * SEQ + t0 + tl) * NPROJ + DTRANK + c];
        }
        for (int tl = lane; tl < SCAN_TS; tl += 32) {
            long long r = (long long)(b * SEQ + t0 + tl) * DI + d;
            sDT[wave][tl] = dt[r];
            sU [wave][tl] = u[r];
        }
        __syncthreads();
        for (int tl = 0; tl < SCAN_TS; ++tl) {
            float sdt = sDT[wave][tl];
            float su  = sU [wave][tl];
            float dA0 = __expf(sdt * A0);
            float dA1 = __expf(sdt * A1);
            float dbu = sdt * su;
            h0 = dA0 * h0 + dbu * sBC[tl][lane];
            h1 = dA1 * h1 + dbu * sBC[tl][32 + lane];
            float acc = h0 * sBC[tl][64 + lane] + h1 * sBC[tl][96 + lane];
            #pragma unroll
            for (int off = 16; off > 0; off >>= 1)
                acc += __shfl_xor(acc, off, 32);
            if (lane == 0)
                y[(long long)(b * SEQ + t0 + tl) * DI + d] = acc;
        }
        __syncthreads();
    }
}

// ---------------- g = (y + u*D) * silu(z) -> bf16 ----------------
__global__ void gate_kernel(const float* __restrict__ y, const float* __restrict__ u,
                            const float* __restrict__ Dp, const float* __restrict__ xz,
                            __bf16* __restrict__ g_bf) {
    long long i = (long long)blockIdx.x * 256 + threadIdx.x;   // over BT*DI
    if (i >= (long long)BT * DI) return;
    int d       = (int)(i % DI);
    long long r = i / DI;
    float z = xz[r * NXZ + DI + d];
    float g = (y[i] + u[i] * Dp[d]) * (z / (1.0f + __expf(-z)));
    g_bf[i] = (__bf16)g;
}

// ---------------- launcher ----------------
extern "C" void kernel_launch(void* const* d_in, const int* in_sizes, int n_in,
                              void* d_out, int out_size, void* d_ws, size_t ws_size,
                              hipStream_t stream) {
    const float*         x      = (const float*)d_in[0];
    const unsigned char* mask   = (const unsigned char*)d_in[1];
    const float*         W_in   = (const float*)d_in[2];
    const float*         conv_w = (const float*)d_in[3];
    const float*         conv_b = (const float*)d_in[4];
    const float*         W_xprj = (const float*)d_in[5];
    const float*         W_dt   = (const float*)d_in[6];
    const float*         b_dt   = (const float*)d_in[7];
    const float*         A_log  = (const float*)d_in[8];
    const float*         Dp     = (const float*)d_in[9];
    const float*         W_out  = (const float*)d_in[10];
    float* out = (float*)d_out;

    // workspace carve-out (256B aligned slices)
    char* w = (char*)d_ws;
    auto alloc = [&](long long bytes) {
        char* p = w; w += (bytes + 255) & ~255LL; return p;
    };
    __bf16* x_bf    = (__bf16*)alloc((long long)BT * DM * 2);
    __bf16* Win_bf  = (__bf16*)alloc((long long)DM * NXZ * 2);
    __bf16* Wxp_bf  = (__bf16*)alloc((long long)DI * NPROJ * 2);
    __bf16* Wout_bf = (__bf16*)alloc((long long)DI * DM * 2);
    float*  xz      = (float*) alloc((long long)BT * NXZ * 4);
    float*  u       = (float*) alloc((long long)BT * DI * 4);
    __bf16* u_bf    = (__bf16*)alloc((long long)BT * DI * 2);
    float*  proj    = (float*) alloc((long long)BT * NPROJ * 4);
    float*  dtbuf   = (float*) alloc((long long)BT * DI * 4);
    float*  ybuf    = (float*) alloc((long long)BT * DI * 4);
    __bf16* g_bf    = (__bf16*)alloc((long long)BT * DI * 2);
    (void)ws_size; (void)in_sizes; (void)n_in; (void)out_size;

    // 1) bf16 conversions (x gets the mask applied)
    {
        long long n = (long long)BT * DM;
        x_mask_to_bf16<<<dim3((unsigned)((n + 255) / 256)), 256, 0, stream>>>(x, mask, x_bf, n);
        n = (long long)DM * NXZ;
        f32_to_bf16<<<dim3((unsigned)((n + 255) / 256)), 256, 0, stream>>>(W_in, Win_bf, n);
        n = (long long)DI * NPROJ;
        f32_to_bf16<<<dim3((unsigned)((n + 255) / 256)), 256, 0, stream>>>(W_xprj, Wxp_bf, n);
        n = (long long)DI * DM;
        f32_to_bf16<<<dim3((unsigned)((n + 255) / 256)), 256, 0, stream>>>(W_out, Wout_bf, n);
    }

    // 2) xz = x @ W_in   (16384 x 768 x 3072)
    gemm_bf16_wmma<<<dim3(NXZ / 64, BT / 128), 256, 0, stream>>>(x_bf, Win_bf, xz, BT, NXZ, DM);

    // 3) depthwise conv + SiLU -> u (f32 + bf16)
    {
        long long n = (long long)BT * DI;
        conv_silu_kernel<<<dim3((unsigned)((n + 255) / 256)), 256, 0, stream>>>(xz, conv_w, conv_b, u, u_bf);
    }

    // 4) proj = u @ W_xproj   (16384 x 1536 x 176)
    gemm_bf16_wmma<<<dim3((NPROJ + 63) / 64, BT / 128), 256, 0, stream>>>(u_bf, Wxp_bf, proj, BT, NPROJ, DI);

    // 5) dt = softplus(proj[:, :48] @ W_dt + b_dt)
    dtproj_kernel<<<dim3(DI / 256, BT), 256, 0, stream>>>(proj, W_dt, b_dt, dtbuf);

    // 6) selective scan -> ybuf
    scan_kernel<<<dim3(DI / 8, BATCH), 256, 0, stream>>>(proj, dtbuf, u, A_log, ybuf);

    // 7) gate: g = (y + u*D) * silu(z)
    {
        long long n = (long long)BT * DI;
        gate_kernel<<<dim3((unsigned)((n + 255) / 256)), 256, 0, stream>>>(ybuf, u, Dp, xz, g_bf);
    }

    // 8) out = g @ W_out   (16384 x 1536 x 768)
    gemm_bf16_wmma<<<dim3(DM / 64, BT / 128), 256, 0, stream>>>(g_bf, Wout_bf, out, BT, DM, DI);
}
